// MultiHeadedAttention_54812372631840
// MI455X (gfx1250) — compile-verified, matrix operand order checked by reading, not run on hardware
//
#include <hip/hip_runtime.h>

#define EMB   768
#define NH    12
#define HD    64
#define BATCH 32
#define SEQ   512
#define BS    (BATCH * SEQ)   // 16384 rows

typedef __attribute__((ext_vector_type(16))) __bf16       v16bf;
typedef __attribute__((ext_vector_type(8)))  float        v8f;
typedef __attribute__((ext_vector_type(4)))  unsigned int u32x4;

union FragU { u32x4 u[2]; v16bf v; };

__device__ __forceinline__ unsigned short f32_to_bf16(float f) {
  unsigned int u = __float_as_uint(f);
  u += 0x7FFFu + ((u >> 16) & 1u);          // round-to-nearest-even
  return (unsigned short)(u >> 16);
}

// Load a 16x32 bf16 A-fragment (or symmetric B-fragment from a [N][K] buffer).
// ISA layout: lane l -> row (l&15); k-chunks {kb..kb+7, kb+16..kb+23}, kb = 8*(l>=16).
__device__ __forceinline__ v16bf load_frag(const unsigned short* base, int stride, int lane) {
  const int r  = lane & 15;
  const int kb = (lane >> 4) << 3;
  const unsigned short* p = base + r * stride + kb;
  FragU f;
  f.u[0] = *reinterpret_cast<const u32x4*>(p);
  f.u[1] = *reinterpret_cast<const u32x4*>(p + 16);
  return f.v;
}

__device__ __forceinline__ v8f wmma_bf16(v16bf a, v16bf b, v8f c) {
  return __builtin_amdgcn_wmma_f32_16x16x32_bf16(
      /*neg_a=*/false, a, /*neg_b=*/false, b,
      /*c_mod=*/(short)0, c, /*reuse_a=*/false, /*reuse_b=*/false);
}

// ---------------- fp32 -> bf16 conversion kernels ----------------

// Packed: 2 floats -> 1 dword of bf16 per thread.
__global__ void cvt_bf16x2(const float2* __restrict__ in, unsigned int* __restrict__ out, int n2) {
  int i = blockIdx.x * blockDim.x + threadIdx.x;
  if (i < n2) {
    float2 v = in[i];
    out[i] = (unsigned int)f32_to_bf16(v.x) | ((unsigned int)f32_to_bf16(v.y) << 16);
  }
}

// W is [K=768][N=768] row-major; store WT[n][k] so B-fragment columns are contiguous.
__global__ void cvt_wT(const float* __restrict__ W, unsigned short* __restrict__ WT) {
  int i = blockIdx.x * blockDim.x + threadIdx.x;   // over 768*768
  int k = i / EMB, n = i % EMB;
  WT[(size_t)n * EMB + k] = f32_to_bf16(W[i]);
}

// ---------------- generic WMMA GEMM: C[M,768] = A[M,768] @ WT^T + bias ----------------
// Each wave computes a 32x64 tile (2 A-frags share 4 B-frags -> 8 WMMA per k-step).
// k-loop unrolled x2 so next iteration's loads overlap this iteration's WMMAs.
// MODE 0: scatter bf16 into Q/K layout [B,H,S,D]
// MODE 2: scatter bf16 into Vt layout  [B,H,D,S]
// MODE 3: fp32 row-major out [M,768]
template <int MODE>
__global__ void gemm_bf16(const unsigned short* __restrict__ A,
                          const unsigned short* __restrict__ BT,
                          const float* __restrict__ bias,
                          void* __restrict__ out) {
  const int lane = threadIdx.x & 31;
  const int wave = threadIdx.x >> 5;
  const int m0 = blockIdx.x * 128 + wave * 32;   // 2 x 16 rows per wave
  const int n0 = blockIdx.y * 64;

  v8f acc0[4] = {};
  v8f acc1[4] = {};
  const unsigned short* Ab0 = A + (size_t)m0 * EMB;
  const unsigned short* Ab1 = Ab0 + (size_t)16 * EMB;

#pragma unroll 2
  for (int k0 = 0; k0 < EMB; k0 += 32) {
    __builtin_prefetch(Ab0 + k0 + 256, 0, 0);
    __builtin_prefetch(Ab1 + k0 + 256, 0, 0);
    v16bf a0 = load_frag(Ab0 + k0, EMB, lane);
    v16bf a1 = load_frag(Ab1 + k0, EMB, lane);
#pragma unroll
    for (int nt = 0; nt < 4; ++nt) {
      v16bf b = load_frag(BT + (size_t)(n0 + nt * 16) * EMB + k0, EMB, lane);
      acc0[nt] = wmma_bf16(a0, b, acc0[nt]);
      acc1[nt] = wmma_bf16(a1, b, acc1[nt]);
    }
  }

  const int hi = (lane >> 4) * 8;
#pragma unroll
  for (int mi = 0; mi < 2; ++mi) {
#pragma unroll
    for (int nt = 0; nt < 4; ++nt) {
      const int col = n0 + nt * 16 + (lane & 15);
      const float bv = bias[col];
#pragma unroll
      for (int r = 0; r < 8; ++r) {
        const int row = m0 + mi * 16 + r + hi;
        const float v = (mi ? acc1[nt][r] : acc0[nt][r]) + bv;
        if (MODE == 3) {
          ((float*)out)[(size_t)row * EMB + col] = v;
        } else {
          const int b = row >> 9, s = row & 511;
          const int h = col >> 6, d = col & 63;
          size_t dst;
          if (MODE == 2) dst = ((size_t)(b * NH + h) * HD + d) * SEQ + s;   // [B,H,D,S]
          else           dst = ((size_t)(b * NH + h) * SEQ + s) * HD + d;   // [B,H,S,D]
          ((unsigned short*)out)[dst] = f32_to_bf16(v);
        }
      }
    }
  }
}

// ---------------- causal attention: one wave per (b, h, 16-row q tile) ----------------
__global__ void __launch_bounds__(32)
attn_kernel(const unsigned short* __restrict__ Q,   // [B,H,S,D] bf16
            const unsigned short* __restrict__ K,   // [B,H,S,D] bf16
            const unsigned short* __restrict__ Vt,  // [B,H,D,S] bf16
            unsigned short* __restrict__ AO) {      // [BS, EMB] bf16
  __shared__ float          sS[16][516];  // fp32 scores (padded vs bank conflicts)
  __shared__ unsigned short sP[16][528];  // bf16 probabilities (16B-aligned rows)

  const int lane = threadIdx.x & 31;
  const int idx = blockIdx.x;
  const int qt = idx & 31;          // 32 q-tiles
  const int bh = idx >> 5;
  const int h  = bh % NH;
  const int b  = bh / NH;
  const int q0 = qt * 16;

  const size_t headQK = (size_t)(b * NH + h) * SEQ * HD;
  const unsigned short* Qb = Q + headQK + (size_t)q0 * HD;
  const unsigned short* Kb = K + headQK;
  const unsigned short* Vb = Vt + (size_t)(b * NH + h) * HD * SEQ;

  // ---- phase 1: scores = (Q Kt) * 1/sqrt(D), causal-masked, into LDS ----
  const int hi = (lane >> 4) * 8;
  const int ntMax = qt;                      // tiles with nt*16 > q0+15 fully masked
  for (int nt = 0; nt <= ntMax; ++nt) {
    v8f acc = {};
#pragma unroll
    for (int kk = 0; kk < HD; kk += 32) {
      v16bf a  = load_frag(Qb + kk, HD, lane);
      v16bf bf = load_frag(Kb + (size_t)(nt * 16) * HD + kk, HD, lane);
      acc = wmma_bf16(a, bf, acc);
    }
    const int col = nt * 16 + (lane & 15);
#pragma unroll
    for (int r = 0; r < 8; ++r) {
      const int row = r + hi;
      const int qi = q0 + row;
      sS[row][col] = (col <= qi) ? acc[r] * 0.125f : -__builtin_inff();
    }
  }
  __syncthreads();

  // ---- phase 2: softmax; 2 lanes per row (256 cols each), combine via shfl_xor(16) ----
  {
    const int row = lane & 15;
    const int half = lane >> 4;
    const int qi = q0 + row;
    const int c0 = half * 256;
    int cend = qi - c0 + 1;
    if (cend > 256) cend = 256;
    if (cend < 0) cend = 0;

    float mloc = -__builtin_inff();
    for (int j = 0; j < cend; ++j) mloc = fmaxf(mloc, sS[row][c0 + j]);
    const float M = fmaxf(mloc, __shfl_xor(mloc, 16, 32));

    float sloc = 0.f;
    for (int j = 0; j < cend; ++j) sloc += __expf(sS[row][c0 + j] - M);
    const float sum = sloc + __shfl_xor(sloc, 16, 32);
    const float inv = 1.f / sum;

    for (int j = 0; j < 256; ++j) {
      const float p = (j < cend) ? __expf(sS[row][c0 + j] - M) * inv : 0.f;
      sP[row][c0 + j] = f32_to_bf16(p);
    }
  }
  __syncthreads();

  // ---- phase 3: out = P @ V  (A from LDS, B from Vt), 16x64 accumulator ----
  v8f o[4] = {};
  const int ktMax = (q0 + 15) >> 5;          // skip fully-zero P tiles
  for (int kt = 0; kt <= ktMax; ++kt) {
    const int sk = kt * 32;
    v16bf a = load_frag(&sP[0][sk], 528, lane);
#pragma unroll
    for (int dt = 0; dt < 4; ++dt) {
      v16bf bf = load_frag(Vb + (size_t)(dt * 16) * SEQ + sk, SEQ, lane);
      o[dt] = wmma_bf16(a, bf, o[dt]);
    }
  }

  const int colb = h * HD + (lane & 15);
#pragma unroll
  for (int dt = 0; dt < 4; ++dt)
#pragma unroll
    for (int r = 0; r < 8; ++r) {
      const int row = q0 + r + hi;
      AO[((size_t)b * SEQ + row) * EMB + colb + dt * 16] = f32_to_bf16(o[dt][r]);
    }
}

// ---------------- host launch ----------------
extern "C" void kernel_launch(void* const* d_in, const int* in_sizes, int n_in,
                              void* d_out, int out_size, void* d_ws, size_t ws_size,
                              hipStream_t stream) {
  const float* x  = (const float*)d_in[0];
  const float* Wq = (const float*)d_in[1];
  const float* bq = (const float*)d_in[2];
  const float* Wk = (const float*)d_in[3];
  const float* bk = (const float*)d_in[4];
  const float* Wv = (const float*)d_in[5];
  const float* bv = (const float*)d_in[6];
  const float* Wo = (const float*)d_in[7];
  const float* bo = (const float*)d_in[8];

  size_t off = 0;
  auto take = [&](size_t bytes) -> void* {
    void* p = (char*)d_ws + off;
    off += (bytes + 255) & ~(size_t)255;
    return p;
  };
  unsigned short* xb  = (unsigned short*)take((size_t)BS * EMB * 2);
  unsigned short* WqT = (unsigned short*)take((size_t)EMB * EMB * 2);
  unsigned short* WkT = (unsigned short*)take((size_t)EMB * EMB * 2);
  unsigned short* WvT = (unsigned short*)take((size_t)EMB * EMB * 2);
  unsigned short* WoT = (unsigned short*)take((size_t)EMB * EMB * 2);
  unsigned short* Qb  = (unsigned short*)take((size_t)BS * EMB * 2);
  unsigned short* Kb  = (unsigned short*)take((size_t)BS * EMB * 2);
  unsigned short* Vt  = (unsigned short*)take((size_t)BS * EMB * 2);
  unsigned short* AO  = (unsigned short*)take((size_t)BS * EMB * 2);

  const int n2 = (BS * EMB) / 2;
  cvt_bf16x2<<<n2 / 256, 256, 0, stream>>>((const float2*)x, (unsigned int*)xb, n2);
  cvt_wT<<<(EMB * EMB) / 256, 256, 0, stream>>>(Wq, WqT);
  cvt_wT<<<(EMB * EMB) / 256, 256, 0, stream>>>(Wk, WkT);
  cvt_wT<<<(EMB * EMB) / 256, 256, 0, stream>>>(Wv, WvT);
  cvt_wT<<<(EMB * EMB) / 256, 256, 0, stream>>>(Wo, WoT);

  dim3 gg(BS / 128, EMB / 64);
  gemm_bf16<0><<<gg, 128, 0, stream>>>(xb, WqT, bq, Qb);
  gemm_bf16<0><<<gg, 128, 0, stream>>>(xb, WkT, bk, Kb);
  gemm_bf16<2><<<gg, 128, 0, stream>>>(xb, WvT, bv, Vt);

  attn_kernel<<<BATCH * NH * (SEQ / 16), 32, 0, stream>>>(Qb, Kb, Vt, AO);

  gemm_bf16<3><<<gg, 128, 0, stream>>>(AO, WoT, bo, d_out);
}
